// BiGCNLayer_10471130268014
// MI455X (gfx1250) — matrix operand order repaired and checked by reading, not compiled, not run on hardware
//
#include <hip/hip_runtime.h>

// Problem sizes (fixed by the reference)
#define NN 4096   // nodes
#define HH 512    // span_emb_size
#define RR 3      // relations
#define KK 256    // HH/2
#define MB 32     // rows per workgroup in the adjacency GEMM
#define APAD 17   // u32 stride per adj row in LDS   (gcd(17,64)=1: conflict-free)
#define BPAD 20   // u32 stride per sup column in LDS (80B: 16B-aligned b128 targets)

typedef __attribute__((ext_vector_type(16))) __bf16   bf16x16;
typedef __attribute__((ext_vector_type(8)))  float     floatx8;
typedef __attribute__((ext_vector_type(8)))  unsigned  u32x8;
typedef __attribute__((ext_vector_type(4)))  unsigned  u32x4;

// Pack two f32 into one bf16 pair with a single v_perm_b32 (truncation).
__device__ __forceinline__ unsigned pack_bf16(float a, float b) {
    return __builtin_amdgcn_perm(__builtin_bit_cast(unsigned, b),
                                 __builtin_bit_cast(unsigned, a),
                                 0x07060302u);   // D = {b[31:16], a[31:16]}
}

// D(16x16 f32) = A(16x32 bf16) * B(32x16 bf16) + C
__device__ __forceinline__ floatx8 wmma_bf16(const u32x8& a, const u32x8& b, floatx8 c) {
    bf16x16 av = __builtin_bit_cast(bf16x16, a);
    bf16x16 bv = __builtin_bit_cast(bf16x16, b);
    return __builtin_amdgcn_wmma_f32_16x16x32_bf16(false, av, false, bv,
                                                   (short)0, c, false, false);
}

// CDNA5 async global->LDS copy (ASYNCcnt path), GV addressing mode.
__device__ __forceinline__ void async_copy_b128(unsigned lds_off, const void* gptr) {
    asm volatile("global_load_async_to_lds_b128 %0, %1, off"
                 :: "v"(lds_off), "v"(gptr) : "memory");
}
__device__ __forceinline__ void wait_asynccnt0() {
    asm volatile("s_wait_asynccnt 0x0" ::: "memory");
}

// =====================================================================
// Kernel 1: sup[dir][r] = inps @ W[dir][r]  (4096x512 @ 512x256), bf16 out,
// stored TRANSPOSED: sup_t[((dir*RR+r)*KK + col)*NN + m] for coalesced
// (and 16B-aligned, async-copyable) staging in kernel 2.
// dir: 0 = bw (cols [0,256) of concat), 1 = fw.
// One wave per 16x16 tile; block=128; grid=(NN/16, KK/64, 2*RR).
// =====================================================================
__global__ void __launch_bounds__(128)
sup_gemm_kernel(const float* __restrict__ inps,
                const float* __restrict__ W_fw,
                const float* __restrict__ W_bw,
                unsigned* __restrict__ sup_t32) {
    const int t = threadIdx.x;
    const int lane = t & 31, w = t >> 5;
    const int lm = lane & 15, hf = lane >> 4;
    const int dir = blockIdx.z / RR, r = blockIdx.z % RR;
    const int m0 = blockIdx.x * 16;
    const int col = (blockIdx.y * 4 + w) * 16 + lm;
    const float* W = (dir ? W_fw : W_bw) + (size_t)r * HH * KK;

    floatx8 acc = {};
    for (int kk = 0; kk < HH; kk += 32) {
        u32x8 au, bu;
#pragma unroll
        for (int p = 0; p < 8; ++p) {
            const int kloc = (p >> 2) * 16 + hf * 8 + (p & 3) * 2;   // A layout
            const float2 f = *(const float2*)(inps + (size_t)(m0 + lm) * HH + kk + kloc);
            au[p] = pack_bf16(f.x, f.y);
            const int k0 = (p >> 2) * 16 + hf * 8 + (p & 3) * 2;     // B layout
            bu[p] = pack_bf16(W[(size_t)(kk + k0) * KK + col],
                              W[(size_t)(kk + k0 + 1) * KK + col]);
        }
        acc = wmma_bf16(au, bu, acc);
    }
    // D: VGPR v -> M = m0 + v + 8*hf. Pack 8 bf16 -> 4 u32, one b128 store.
    u32x4 o;
#pragma unroll
    for (int v2 = 0; v2 < 4; ++v2) o[v2] = pack_bf16(acc[2 * v2], acc[2 * v2 + 1]);
    unsigned* dst = sup_t32 + ((size_t)((dir * RR + r) * KK + col)) * (NN / 2)
                            + ((m0 + 8 * hf) >> 1);
    *(u32x4*)dst = o;
}

// =====================================================================
// Kernel 2: acc[:, dir*KK:+KK] = sum_r adj[dir][r] @ sup[dir][r] + bias
// grid=(NN/MB, 2), block=256 (8 waves = 2 m-tiles x 4 col-groups of 64).
// Software pipeline with double-buffered LDS:
//   - sup chunk (256 cols x 32 bf16): global_load_async_to_lds_b128 (ASYNCcnt)
//   - adj tile (32x32 f32): global->VGPR (issued one chunk ahead),
//     v_perm_b32 pack -> LDS
// Adjacency (402 MB, the HBM-bound stream) is read exactly once, coalesced.
// =====================================================================
__global__ void __launch_bounds__(256)
adj_gemm_kernel(const float* __restrict__ fw_adjs,
                const float* __restrict__ bw_adjs,
                const float* __restrict__ b_fw,
                const float* __restrict__ b_bw,
                const unsigned* __restrict__ sup_u32,
                float* __restrict__ accOut) {
    const int t = threadIdx.x;
    const int lane = t & 31, w = t >> 5;
    const int lm = lane & 15, hf = lane >> 4;
    const int m0 = blockIdx.x * MB;
    const int dir = blockIdx.y;                 // 0=bw, 1=fw
    const float* adjs = dir ? fw_adjs : bw_adjs;
    const float* bias = dir ? b_fw : b_bw;
    const int mt = (w >> 2) * 16;               // wave's m-tile within block
    const int c0 = (w & 3) * 64;                // wave's 4 column tiles

    __shared__ unsigned ldsA[2][MB * APAD];     // 32x32 bf16 adj tile (packed pairs)
    __shared__ unsigned ldsB[2][256 * BPAD];    // 256 cols x 16 u32 sup chunk

    const unsigned* supBase = sup_u32 + (size_t)dir * RR * KK * (NN / 2);

    // Per-thread staging geometry.
    const int arow = t >> 4;                    // 0..15 (second set: +16)
    const int apair = t & 15;                   // pair index within row (2 cols)

    floatx8 acc[4] = {{}, {}, {}, {}};
    const int TOT = RR * (NN / 32);             // 384 chunks

    // ---- pipeline helpers ----
    auto issueB = [&](int c, int buf) {
        const int r = c >> 7;
        const int nn = (c & 127) << 5;
        const unsigned* src = supBase + (size_t)r * KK * (NN / 2)
                                      + (size_t)t * (NN / 2) + (nn >> 1);
        const unsigned lds0 = (unsigned)(size_t)&ldsB[buf][t * BPAD];
#pragma unroll
        for (int j = 0; j < 4; ++j)
            async_copy_b128(lds0 + 16u * j, src + 4 * j);
    };
    auto issueA = [&](int c, float2& fa, float2& fb) {
        const int r = c >> 7;
        const int nn = (c & 127) << 5;
        const float* A = adjs + (size_t)r * NN * NN + (size_t)m0 * NN + nn;
        fa = *(const float2*)(A + (size_t)arow * NN + 2 * apair);
        fb = *(const float2*)(A + (size_t)(arow + 16) * NN + 2 * apair);
    };
    auto storeA = [&](int buf, float2 fa, float2 fb) {
        ldsA[buf][arow * APAD + apair]        = pack_bf16(fa.x, fa.y);
        ldsA[buf][(arow + 16) * APAD + apair] = pack_bf16(fb.x, fb.y);
    };

    // ---- prologue: fill buffer 0 ----
    float2 fa, fb;
    issueB(0, 0);
    issueA(0, fa, fb);
    storeA(0, fa, fb);
    wait_asynccnt0();
    __syncthreads();

    // ---- main pipelined loop ----
    for (int c = 0; c < TOT; ++c) {
        const int cur = c & 1, nxt = cur ^ 1;
        const bool more = (c + 1) < TOT;
        if (more) {                              // issue next chunk early
            issueB(c + 1, nxt);
            issueA(c + 1, fa, fb);
        }
        // A fragment (shared by the 4 col-group waves of this m-tile)
        u32x8 au;
#pragma unroll
        for (int p = 0; p < 8; ++p) {
            const int kh = hf * 8 + p;
            au[p] = ldsA[cur][(mt + lm) * APAD + kh];
        }
        // 4 B fragments + 4 WMMAs
#pragma unroll
        for (int ct = 0; ct < 4; ++ct) {
            u32x8 bu;
#pragma unroll
            for (int p = 0; p < 8; ++p)
                bu[p] = ldsB[cur][(c0 + ct * 16 + lm) * BPAD + hf * 8 + p];
            acc[ct] = wmma_bf16(au, bu, acc[ct]);
        }
        if (more) storeA(nxt, fa, fb);
        wait_asynccnt0();
        __syncthreads();
    }

    // ---- bias (summed over relations) + writeout ----
#pragma unroll
    for (int ct = 0; ct < 4; ++ct) {
        const int colL = c0 + ct * 16 + lm;
        float bs = 0.f;
#pragma unroll
        for (int r = 0; r < RR; ++r) bs += bias[r * KK + colL];
#pragma unroll
        for (int v = 0; v < 8; ++v) {
            const int m = m0 + mt + v + 8 * hf;
            accOut[(size_t)m * HH + dir * KK + colL] = acc[ct][v] + bs;
        }
    }
}

// =====================================================================
// Kernel 3: out = relu(acc) @ W1 + b1 + inps  (4096x512 @ 512x512)
// relu + f32->bf16 fused into the A-fragment load.
// grid=(NN/16, HH/64), block=128 (one 16x16 tile per wave).
// =====================================================================
__global__ void __launch_bounds__(128)
final_linear_kernel(const float* __restrict__ accIn,
                    const float* __restrict__ W1,
                    const float* __restrict__ b1,
                    const float* __restrict__ inps,
                    float* __restrict__ out) {
    const int t = threadIdx.x;
    const int lane = t & 31, w = t >> 5;
    const int lm = lane & 15, hf = lane >> 4;
    const int m0 = blockIdx.x * 16;
    const int colg = (blockIdx.y * 4 + w) * 16 + lm;

    floatx8 d = {};
    for (int kk = 0; kk < HH; kk += 32) {
        u32x8 au, bu;
#pragma unroll
        for (int p = 0; p < 8; ++p) {
            const int kloc = (p >> 2) * 16 + hf * 8 + (p & 3) * 2;
            const float2 f = *(const float2*)(accIn + (size_t)(m0 + lm) * HH + kk + kloc);
            au[p] = pack_bf16(fmaxf(f.x, 0.f), fmaxf(f.y, 0.f));
            const int k0 = (p >> 2) * 16 + hf * 8 + (p & 3) * 2;
            bu[p] = pack_bf16(W1[(size_t)(kk + k0) * HH + colg],
                              W1[(size_t)(kk + k0 + 1) * HH + colg]);
        }
        d = wmma_bf16(au, bu, d);
    }
    const float bb = b1[colg];
#pragma unroll
    for (int v = 0; v < 8; ++v) {
        const int m = m0 + v + 8 * hf;
        out[(size_t)m * HH + colg] = d[v] + bb + inps[(size_t)m * HH + colg];
    }
}

// =====================================================================
extern "C" void kernel_launch(void* const* d_in, const int* in_sizes, int n_in,
                              void* d_out, int out_size, void* d_ws, size_t ws_size,
                              hipStream_t stream) {
    const float* inps    = (const float*)d_in[0];
    const float* fw_adjs = (const float*)d_in[1];
    const float* bw_adjs = (const float*)d_in[2];
    const float* W_fw    = (const float*)d_in[3];
    const float* b_fw    = (const float*)d_in[4];
    const float* W_bw    = (const float*)d_in[5];
    const float* b_bw    = (const float*)d_in[6];
    const float* W1      = (const float*)d_in[7];
    const float* b1      = (const float*)d_in[8];
    float* out = (float*)d_out;

    // Workspace: sup_t (bf16, transposed) 12.58 MB, then acc (f32) 8.39 MB.
    unsigned* sup_t32 = (unsigned*)d_ws;
    const size_t sup_bytes = (size_t)2 * RR * KK * NN * 2;
    float* acc = (float*)((char*)d_ws + sup_bytes);

    sup_gemm_kernel<<<dim3(NN / 16, KK / 64, 2 * RR), 128, 0, stream>>>(
        inps, W_fw, W_bw, sup_t32);

    adj_gemm_kernel<<<dim3(NN / MB, 2), 256, 0, stream>>>(
        fw_adjs, bw_adjs, b_fw, b_bw, (const unsigned*)sup_t32, acc);

    final_linear_kernel<<<dim3(NN / 16, HH / 64), 128, 0, stream>>>(
        acc, W1, b1, inps, out);
}
